// IMG2TEXT_16982300689128
// MI455X (gfx1250) — matrix operand order repaired and checked by reading, not compiled, not run on hardware
//
#include <hip/hip_runtime.h>
#include <hip/hip_bf16.h>

// ---------------------------------------------------------------------------
// CDNA5 (gfx1250) implementation of the 2-layer template-token transformer.
// All GEMMs go through v_wmma_f32_16x16x32_bf16 (wave32 WMMA), LDS-staged.
//
// Input order (jax tree-flatten: dict keys sorted alphabetically):
//   0  image_patch_feats (128,257,1024) f32
//   1  img_global_feat   (128,768)      f32
//   2  fc_b (768)   3 fc_w (1024,768)
//   4..17  layer0: b1,b2,bkv,bproj,bq,ln1_b,ln1_s,ln2_b,ln2_s,w1,w2,wkv,wproj,wq
//   18..31 layer1: same order
//   32 templates (1,16,1024)
// Output: sel (128,8,768) f32 followed by num_r (128) stored as float.
// ---------------------------------------------------------------------------

typedef unsigned short bf16s;  // raw bf16 storage
typedef __attribute__((ext_vector_type(16))) __bf16 v16bf;
typedef __attribute__((ext_vector_type(8)))  float  v8f;

union FragBF { v16bf v; unsigned int u[8]; };

#define TOKENS   34944     // 128 * 273
#define NTOK     273
#define NPAD     288
#define DMODEL   1024
#define DHEAD    128
#define NHEAD    8
#define BATCH    128

static __device__ __forceinline__ bf16s f2bf(float f) {
  unsigned int u = __float_as_uint(f);
  unsigned int r = (u + 0x7FFFu + ((u >> 16) & 1u)) >> 16;
  return (bf16s)r;
}

// A fragment (16x32 bf16), ISA 7.12.2: VGPR r, lane half: k0=(r/4)*16+half*8+(r%4)*2
static __device__ __forceinline__ v16bf lds_frag_a(const bf16s* tile, int m, int half) {
  FragBF f;
  const unsigned int* p = (const unsigned int*)(tile + m * 32);
#pragma unroll
  for (int r = 0; r < 8; ++r) {
    int k0 = ((r >> 2) << 4) + (half << 3) + ((r & 3) << 1);
    f.u[r] = p[k0 >> 1];
  }
  return f.v;
}

// B fragment (32x16 bf16) from LDS tile stored transposed [n][k]:
// VGPR r holds K = (r/4)*16 + half*8 + (r%4)*2, +1 for column n = lane%16
static __device__ __forceinline__ v16bf lds_frag_b(const bf16s* tileT, int n, int half) {
  FragBF f;
  const unsigned int* p = (const unsigned int*)(tileT + n * 32);
#pragma unroll
  for (int r = 0; r < 8; ++r) f.u[r] = p[((r >> 2) << 3) + (half << 2) + (r & 3)];
  return f.v;
}

// ---------------------------------------------------------------------------
// 0) Build x = concat(templates, patches) in f32
// ---------------------------------------------------------------------------
__global__ void concat_kernel(const float* __restrict__ patches,
                              const float* __restrict__ tpl,
                              float* __restrict__ x) {
  long idx = (long)blockIdx.x * 256 + threadIdx.x;   // TOKENS*DMODEL threads
  int d = (int)(idx & 1023);
  long t = idx >> 10;
  int n = (int)(t % NTOK);
  long b = t / NTOK;
  x[idx] = (n < 16) ? tpl[n * DMODEL + d]
                    : patches[(b * 257 + (n - 16)) * DMODEL + d];
}

// ---------------------------------------------------------------------------
// 1) LayerNorm over D=1024 per token, output bf16
// ---------------------------------------------------------------------------
__global__ __launch_bounds__(256)
void ln_kernel(const float* __restrict__ x, const float* __restrict__ gamma,
               const float* __restrict__ beta, bf16s* __restrict__ out) {
  int tok = blockIdx.x;
  int tid = threadIdx.x;
  const float* row = x + (long)tok * DMODEL;
  __shared__ float red[256];
  float v[4];
  float lsum = 0.f;
#pragma unroll
  for (int i = 0; i < 4; ++i) { v[i] = row[tid + 256 * i]; lsum += v[i]; }
  red[tid] = lsum; __syncthreads();
  for (int s = 128; s > 0; s >>= 1) { if (tid < s) red[tid] += red[tid + s]; __syncthreads(); }
  float mean = red[0] * (1.0f / DMODEL); __syncthreads();
  float lvar = 0.f;
#pragma unroll
  for (int i = 0; i < 4; ++i) { float d = v[i] - mean; lvar += d * d; }
  red[tid] = lvar; __syncthreads();
  for (int s = 128; s > 0; s >>= 1) { if (tid < s) red[tid] += red[tid + s]; __syncthreads(); }
  float rstd = rsqrtf(red[0] * (1.0f / DMODEL) + 1e-5f);
#pragma unroll
  for (int i = 0; i < 4; ++i) {
    int d = tid + 256 * i;
    out[(long)tok * DMODEL + d] = f2bf((v[i] - mean) * rstd * gamma[d] + beta[d]);
  }
}

// ---------------------------------------------------------------------------
// 2) Generic WMMA GEMM: C(MxN) = A_bf16(MxK) @ W_f32(KxN) + bias [+resid] [act]
//    act: 0 none, 1 relu, 2 sigmoid.  outB!=null -> bf16 output else f32.
//    M,N multiples of 64, K multiple of 32.
//    Block = 256 threads (8 waves) -> 64x64 C tile; wave w computes a 16x32
//    strip (rows (w&3)*16, cols (w>>2)*32): 2 accumulators sharing one A-frag.
//    Double-buffered LDS: stage tile t+1 while WMMAs consume tile t.
// ---------------------------------------------------------------------------
__global__ __launch_bounds__(256)
void gemm_kernel(const bf16s* __restrict__ A, const float* __restrict__ W,
                 const float* __restrict__ bias, const float* __restrict__ resid,
                 float* __restrict__ outF, bf16s* __restrict__ outB,
                 int M, int N, int K, int act) {
  __shared__ __attribute__((aligned(16))) bf16s aT[2][64 * 32];  // [i][k]
  __shared__ __attribute__((aligned(16))) bf16s bT[2][64 * 32];  // [n][k]
  int tid = threadIdx.x, lane = tid & 31, wave = tid >> 5;
  int wm = wave & 3, wn = wave >> 2;
  long rowb = (long)blockIdx.y * 64;
  int colb = blockIdx.x * 64;
  int nkt = K >> 5;
  v8f acc[2] = {};

  // prologue: stage K-tile 0
  for (int e = tid; e < 2048; e += 256) {
    int i = e >> 5, k = e & 31;
    aT[0][e] = A[(rowb + i) * (long)K + k];
  }
  for (int e = tid; e < 2048; e += 256) {
    int k = e >> 6, n = e & 63;
    bT[0][n * 32 + k] = f2bf(W[(long)k * N + colb + n]);
  }

  for (int t = 0; t < nkt; ++t) {
    int cur = t & 1;
    __syncthreads();   // staging of buf[cur] done; reads of buf[cur^1] done
    if (t + 1 < nkt) { // stage next tile into the other buffer
      int kb = (t + 1) << 5;
      if (t + 2 < nkt) {  // global_prefetch_b8 two tiles ahead
        __builtin_prefetch(&A[(rowb + (tid >> 2)) * (long)K + ((t + 2) << 5) + ((tid & 3) << 3)], 0, 1);
        __builtin_prefetch(&W[(long)(((t + 2) << 5) + (tid >> 3)) * N + colb + ((tid & 7) << 3)], 0, 1);
      }
      for (int e = tid; e < 2048; e += 256) {
        int i = e >> 5, k = e & 31;
        aT[cur ^ 1][e] = A[(rowb + i) * (long)K + kb + k];
      }
      for (int e = tid; e < 2048; e += 256) {
        int k = e >> 6, n = e & 63;
        bT[cur ^ 1][n * 32 + k] = f2bf(W[(long)(kb + k) * N + colb + n]);
      }
    }
    int half = lane >> 4;
    v16bf fa  = lds_frag_a(aT[cur], wm * 16 + (lane & 15), half);
    v16bf fb0 = lds_frag_b(bT[cur], wn * 32 + (lane & 15), half);
    v16bf fb1 = lds_frag_b(bT[cur], wn * 32 + 16 + (lane & 15), half);
    acc[0] = __builtin_amdgcn_wmma_f32_16x16x32_bf16(false, fa, false, fb0,
                                                     (short)0, acc[0], false, false);
    acc[1] = __builtin_amdgcn_wmma_f32_16x16x32_bf16(false, fa, false, fb1,
                                                     (short)0, acc[1], false, false);
  }
  int n = lane & 15, half = lane >> 4;
#pragma unroll
  for (int j = 0; j < 2; ++j) {
    int col = colb + wn * 32 + j * 16 + n;
    float bv = bias ? bias[col] : 0.0f;
#pragma unroll
    for (int r = 0; r < 8; ++r) {
      long row = rowb + wm * 16 + r + 8 * half;
      float val = acc[j][r] + bv;
      if (resid) val += resid[row * N + col];
      if (act == 1) val = val > 0.f ? val : 0.f;
      else if (act == 2) val = 1.0f / (1.0f + __expf(-val));
      if (outB) outB[row * (long)N + col] = f2bf(val);
      else outF[row * (long)N + col] = val;
    }
  }
}

// ---------------------------------------------------------------------------
// 3) Attention scores S[b,h] = Q Kt * dh^-0.5, padded to 288x288
// ---------------------------------------------------------------------------
__global__ __launch_bounds__(128)
void attn_scores_kernel(const bf16s* __restrict__ qb, const bf16s* __restrict__ kvb,
                        float* __restrict__ s) {
  int bh = blockIdx.z, b = bh >> 3, h = bh & 7;
  int mb = blockIdx.y * 32, nb = blockIdx.x * 32;
  __shared__ __attribute__((aligned(16))) bf16s aT[32 * 32];
  __shared__ __attribute__((aligned(16))) bf16s bT[32 * 32];
  int tid = threadIdx.x, lane = tid & 31, wave = tid >> 5;
  int wm = wave >> 1, wn = wave & 1;
  v8f acc = {};
  for (int kb = 0; kb < DHEAD; kb += 32) {
    for (int e = tid; e < 1024; e += 128) {
      int i = e >> 5, k = e & 31, qi = mb + i;
      aT[e] = (qi < NTOK) ? qb[((long)b * NTOK + qi) * DMODEL + h * DHEAD + kb + k] : (bf16s)0;
    }
    for (int e = tid; e < 1024; e += 128) {
      int i = e >> 5, k = e & 31, kj = nb + i;   // bT[j][k] = K[j][kb+k]
      bT[i * 32 + k] = (kj < NTOK) ? kvb[((long)b * NTOK + kj) * 2048 + h * DHEAD + kb + k] : (bf16s)0;
    }
    __syncthreads();
    int half = lane >> 4;
    v16bf fa = lds_frag_a(aT, wm * 16 + (lane & 15), half);
    v16bf fb = lds_frag_b(bT, wn * 16 + (lane & 15), half);
    acc = __builtin_amdgcn_wmma_f32_16x16x32_bf16(false, fa, false, fb,
                                                  (short)0, acc, false, false);
    __syncthreads();
  }
  const float scale = 0.08838834764831845f;  // 128^-0.5
  int n = lane & 15, half = lane >> 4;
#pragma unroll
  for (int r = 0; r < 8; ++r) {
    int row = mb + wm * 16 + r + 8 * half;
    s[((long)bh * NPAD + row) * NPAD + nb + wn * 16 + n] = acc[r] * scale;
  }
}

// ---------------------------------------------------------------------------
// 4) Softmax over 273 keys -> bf16 probs, zero-filled pad columns
// ---------------------------------------------------------------------------
__global__ __launch_bounds__(256)
void softmax_kernel(const float* __restrict__ s, bf16s* __restrict__ p) {
  int i = blockIdx.x, h = blockIdx.y, b = blockIdx.z;
  long base = (((long)b * NHEAD + h) * NPAD + i) * NPAD;
  int tid = threadIdx.x;
  float v0 = (tid < NTOK) ? s[base + tid] : -3.0e38f;
  float v1 = (tid + 256 < NTOK) ? s[base + tid + 256] : -3.0e38f;
  __shared__ float red[256];
  red[tid] = fmaxf(v0, v1); __syncthreads();
  for (int st = 128; st > 0; st >>= 1) { if (tid < st) red[tid] = fmaxf(red[tid], red[tid + st]); __syncthreads(); }
  float mx = red[0]; __syncthreads();
  float e0 = (tid < NTOK) ? __expf(v0 - mx) : 0.f;
  float e1 = (tid + 256 < NTOK) ? __expf(v1 - mx) : 0.f;
  red[tid] = e0 + e1; __syncthreads();
  for (int st = 128; st > 0; st >>= 1) { if (tid < st) red[tid] += red[tid + st]; __syncthreads(); }
  float inv = 1.0f / red[0];
  p[base + tid] = f2bf(e0 * inv);
  if (tid + 256 < NPAD) p[base + tid + 256] = f2bf(e1 * inv);
}

// ---------------------------------------------------------------------------
// 5) O[b,h] = P @ V, output bf16 in (b, n, h*dh) layout
// ---------------------------------------------------------------------------
__global__ __launch_bounds__(128)
void attn_av_kernel(const bf16s* __restrict__ p, const bf16s* __restrict__ kvb,
                    bf16s* __restrict__ o) {
  int bh = blockIdx.z, b = bh >> 3, h = bh & 7;
  int mb = blockIdx.y * 32, nb = blockIdx.x * 32;   // nb within dh
  __shared__ __attribute__((aligned(16))) bf16s aT[32 * 32];
  __shared__ __attribute__((aligned(16))) bf16s bT[32 * 32];
  int tid = threadIdx.x, lane = tid & 31, wave = tid >> 5;
  int wm = wave >> 1, wn = wave & 1;
  v8f acc = {};
  for (int kb = 0; kb < NPAD; kb += 32) {
    for (int e = tid; e < 1024; e += 128) {
      int i = e >> 5, k = e & 31;
      aT[e] = p[((long)bh * NPAD + mb + i) * NPAD + kb + k];
    }
    for (int e = tid; e < 1024; e += 128) {
      int k = e >> 5, n = e & 31, key = kb + k;    // bT[n][k] = V[key][nb+n]
      bT[n * 32 + k] = (key < NTOK)
          ? kvb[((long)b * NTOK + key) * 2048 + 1024 + h * DHEAD + nb + n] : (bf16s)0;
    }
    __syncthreads();
    int half = lane >> 4;
    v16bf fa = lds_frag_a(aT, wm * 16 + (lane & 15), half);
    v16bf fb = lds_frag_b(bT, wn * 16 + (lane & 15), half);
    acc = __builtin_amdgcn_wmma_f32_16x16x32_bf16(false, fa, false, fb,
                                                  (short)0, acc, false, false);
    __syncthreads();
  }
  int n = lane & 15, half = lane >> 4;
#pragma unroll
  for (int r = 0; r < 8; ++r) {
    int qi = mb + wm * 16 + r + 8 * half;
    if (qi < NTOK)
      o[((long)b * NTOK + qi) * DMODEL + h * DHEAD + nb + wn * 16 + n] = f2bf(acc[r]);
  }
}

// ---------------------------------------------------------------------------
// 6) Gather first 16 tokens per batch -> bf16 (2048 x 1024)
// ---------------------------------------------------------------------------
__global__ void gather16_kernel(const float* __restrict__ x, bf16s* __restrict__ out) {
  int idx = blockIdx.x * 256 + threadIdx.x;  // 2048*1024
  int d = idx & 1023, t = idx >> 10, b = t >> 4, k = t & 15;
  out[idx] = f2bf(x[((long)b * NTOK + k) * DMODEL + d]);
}

// ---------------------------------------------------------------------------
// 7) Per-batch epilogue: aw softmax, top-8 (stable ties), select_id, normalize
// ---------------------------------------------------------------------------
__global__ __launch_bounds__(32)
void select_kernel(const float* __restrict__ latent, const float* __restrict__ g,
                   float* __restrict__ out) {
  int b = blockIdx.x, lane = threadIdx.x;
  __shared__ float aw[16];
  const float* grow = g + (long)b * 768;
  for (int k = 0; k < 16; ++k) {
    const float* lrow = latent + ((long)b * 16 + k) * 768;
    float s = 0.f;
    for (int e = lane; e < 768; e += 32) s += lrow[e] * grow[e];
#pragma unroll
    for (int off = 16; off > 0; off >>= 1) s += __shfl_xor(s, off, 32);
    if (lane == 0) aw[k] = s;
  }
  __syncthreads();
  // scalar logic replicated on all 32 lanes (deterministic)
  float mx = -3.0e38f;
  for (int k = 0; k < 16; ++k) mx = fmaxf(mx, aw[k]);
  float prob[16]; float sum = 0.f;
  for (int k = 0; k < 16; ++k) { prob[k] = __expf(aw[k] - mx); sum += prob[k]; }
  float inv = 1.0f / sum;
  int cnt = 0;
  for (int k = 0; k < 16; ++k) { prob[k] *= inv; if (prob[k] > 0.07f) cnt++; }
  int numr = cnt < 1 ? 1 : (cnt > 8 ? 8 : cnt);
  int idx[8]; bool used[16];
  for (int k = 0; k < 16; ++k) used[k] = false;
  for (int pos = 0; pos < 8; ++pos) {          // top-k, ties -> lower index
    int best = 0; float bv = -3.0e38f;
    for (int k = 0; k < 16; ++k)
      if (!used[k] && prob[k] > bv) { bv = prob[k]; best = k; }
    used[best] = true; idx[pos] = best;
  }
  int sel[8];
  for (int pos = 0; pos < 8; ++pos) sel[pos] = idx[pos];
  for (int a = 0; a < numr; ++a)               // sort head ascending
    for (int c = a + 1; c < numr; ++c)
      if (sel[c] < sel[a]) { int t = sel[a]; sel[a] = sel[c]; sel[c] = t; }
  for (int pos = 0; pos < 8; ++pos) {
    const float* lrow = latent + ((long)b * 16 + sel[pos]) * 768;
    float s = 0.f;
    for (int e = lane; e < 768; e += 32) { float t = lrow[e]; s += t * t; }
#pragma unroll
    for (int off = 16; off > 0; off >>= 1) s += __shfl_xor(s, off, 32);
    float nrm = sqrtf(s);
    float ninv = 1.0f / fmaxf(nrm, 1e-12f);
    for (int e = lane; e < 768; e += 32)
      out[((long)b * 8 + pos) * 768 + e] = lrow[e] * ninv;
  }
  if (lane == 0) out[(long)BATCH * 8 * 768 + b] = (float)numr;
}

// ---------------------------------------------------------------------------
extern "C" void kernel_launch(void* const* d_in, const int* in_sizes, int n_in,
                              void* d_out, int out_size, void* d_ws, size_t ws_size,
                              hipStream_t stream) {
  (void)in_sizes; (void)n_in; (void)out_size; (void)ws_size;
  const float* patches = (const float*)d_in[0];
  const float* gfeat   = (const float*)d_in[1];
  const float* fc_b    = (const float*)d_in[2];
  const float* fc_w    = (const float*)d_in[3];
  const float* tpl     = (const float*)d_in[32];
  // per-layer params (alphabetical): 0 b1, 1 b2, 2 bkv, 3 bproj, 4 bq,
  // 5 ln1_b, 6 ln1_s, 7 ln2_b, 8 ln2_s, 9 w1, 10 w2, 11 wkv, 12 wproj, 13 wq
  auto LP = [&](int l, int j) { return (const float*)d_in[4 + l * 14 + j]; };

  char* ws = (char*)d_ws;
  size_t off = 0;
  auto carve = [&](size_t bytes) { void* p = ws + off; off += (bytes + 255) & ~(size_t)255; return p; };
  float* x      = (float*)carve((size_t)TOKENS * DMODEL * 4);
  bf16s* h      = (bf16s*)carve((size_t)TOKENS * DMODEL * 2);   // LN out, reused for O
  bf16s* q      = (bf16s*)carve((size_t)TOKENS * DMODEL * 2);
  bf16s* kv     = (bf16s*)carve((size_t)TOKENS * 2048 * 2);     // reused for MLP hidden
  float* s      = (float*)carve((size_t)BATCH * NHEAD * NPAD * NPAD * 4);
  bf16s* p      = (bf16s*)carve((size_t)BATCH * NHEAD * NPAD * NPAD * 2);
  bf16s* lin    = (bf16s*)carve((size_t)2048 * DMODEL * 2);
  float* latent = (float*)carve((size_t)2048 * 768 * 4);

  concat_kernel<<<TOKENS * DMODEL / 256, 256, 0, stream>>>(patches, tpl, x);

  for (int l = 0; l < 2; ++l) {
    ln_kernel<<<TOKENS, 256, 0, stream>>>(x, LP(l, 6), LP(l, 5), h);
    // Q = h @ wq + bq  (bf16 out)
    gemm_kernel<<<dim3(DMODEL / 64, TOKENS / 64), 256, 0, stream>>>(
        h, LP(l, 13), LP(l, 4), nullptr, nullptr, q, TOKENS, DMODEL, DMODEL, 0);
    // KV = h @ wkv + bkv (bf16 out)
    gemm_kernel<<<dim3(2048 / 64, TOKENS / 64), 256, 0, stream>>>(
        h, LP(l, 11), LP(l, 2), nullptr, nullptr, kv, TOKENS, 2048, DMODEL, 0);
    attn_scores_kernel<<<dim3(NPAD / 32, NPAD / 32, BATCH * NHEAD), 128, 0, stream>>>(q, kv, s);
    softmax_kernel<<<dim3(NTOK, NHEAD, BATCH), 256, 0, stream>>>(s, p);
    attn_av_kernel<<<dim3(DHEAD / 32, NPAD / 32, BATCH * NHEAD), 128, 0, stream>>>(p, kv, h);
    // x = x + O @ wproj + bproj (f32 out, residual)
    gemm_kernel<<<dim3(DMODEL / 64, TOKENS / 64), 256, 0, stream>>>(
        h, LP(l, 12), LP(l, 3), x, x, nullptr, TOKENS, DMODEL, DMODEL, 0);
    ln_kernel<<<TOKENS, 256, 0, stream>>>(x, LP(l, 8), LP(l, 7), h);
    // m1 = relu(h2 @ w1 + b1) (bf16 out, into kv buffer)
    gemm_kernel<<<dim3(2048 / 64, TOKENS / 64), 256, 0, stream>>>(
        h, LP(l, 9), LP(l, 0), nullptr, nullptr, kv, TOKENS, 2048, DMODEL, 1);
    // x = x + m1 @ w2 + b2
    gemm_kernel<<<dim3(DMODEL / 64, TOKENS / 64), 256, 0, stream>>>(
        kv, LP(l, 10), LP(l, 1), x, x, nullptr, TOKENS, DMODEL, 2048, 0);
  }

  gather16_kernel<<<2048 * DMODEL / 256, 256, 0, stream>>>(x, lin);
  // latent = sigmoid(x16 @ fc_w + fc_b)  (f32 out)
  gemm_kernel<<<dim3(768 / 64, 2048 / 64), 256, 0, stream>>>(
      lin, fc_w, fc_b, nullptr, latent, nullptr, 2048, 768, DMODEL, 2);
  select_kernel<<<BATCH, 32, 0, stream>>>(latent, gfeat, (float*)d_out);
}
